// SemaEmb_42717744726288
// MI455X (gfx1250) — matrix operand order repaired and checked by reading, not compiled, not run on hardware
//
#include <hip/hip_runtime.h>
#include <hip/hip_bf16.h>

typedef __attribute__((ext_vector_type(16))) _Float16 v16h;
typedef __attribute__((ext_vector_type(8)))  float    v8f;

#define EMB 128
#define PDIM 4

// leaky_relu(x) == max(x, 0.01*x) exactly -> v_mul + v_max_num_f32, no VCC dep
__device__ __forceinline__ float lrelu_f(float x) { return fmaxf(x, 0.01f * x); }

// ---------------------------------------------------------------------------
// zero fill
__global__ void zero_kernel(float* __restrict__ p, long n) {
    long i = (long)blockIdx.x * blockDim.x + threadIdx.x;
    if (i < n) p[i] = 0.0f;
}

// ---------------------------------------------------------------------------
// degree count: deg[idx[e]] += 1
__global__ void count_deg_kernel(const int* __restrict__ idx, float* __restrict__ deg, int E) {
    int e = blockIdx.x * blockDim.x + threadIdx.x;
    if (e < E) atomicAdd(deg + idx[e], 1.0f);
}

// ---------------------------------------------------------------------------
// scatter add: agg[dst[e]] += feat[src[e]] ; 32 threads per edge, float4 each
__global__ void scatter_add_kernel(const float* __restrict__ feat,
                                   const int* __restrict__ src,
                                   const int* __restrict__ dst,
                                   float* __restrict__ agg, int E) {
    long t = (long)blockIdx.x * blockDim.x + threadIdx.x;
    int e = (int)(t >> 5);
    if (e >= E) return;
    int c4 = (int)(t & 31) * 4;
    int s = src[e], d = dst[e];
    const float4 v = *(const float4*)(feat + (size_t)s * EMB + c4);
    float* o = agg + (size_t)d * EMB + c4;
    atomicAdd(o + 0, v.x);
    atomicAdd(o + 1, v.y);
    atomicAdd(o + 2, v.z);
    atomicAdd(o + 3, v.w);
}

// ---------------------------------------------------------------------------
// T[t][r][j] = table_t[r] . opmlp_w1[t*128 .. t*128+127][j]   (t=0: op table, t=1..4: param table)
__global__ void precompute_T_kernel(const float* __restrict__ op_tab,
                                    const float* __restrict__ par_tab,
                                    const float* __restrict__ W1,
                                    float* __restrict__ T) {
    int idx = blockIdx.x * blockDim.x + threadIdx.x;  // 5*64*128 total
    if (idx >= 5 * 64 * EMB) return;
    int j = idx & 127;
    int row = (idx >> 7) & 63;
    int t = idx >> 13;
    const float* tab = ((t == 0) ? op_tab : par_tab) + (size_t)row * EMB;
    const float* w = W1 + (size_t)t * EMB * EMB;
    float s = 0.0f;
#pragma unroll 8
    for (int k = 0; k < EMB; k++) s += tab[k] * w[k * EMB + j];
    T[idx] = s;
}

// ---------------------------------------------------------------------------
// pre[n][j] = b1[j] + T0[ops[n]][j] + sum_p Tp[params[n][p]][j]
__global__ void gather_pre_kernel(const int* __restrict__ ops,
                                  const int* __restrict__ params,
                                  const float* __restrict__ T,
                                  const float* __restrict__ b1,
                                  float* __restrict__ pre, int N) {
    long idx = (long)blockIdx.x * blockDim.x + threadIdx.x;
    int n = (int)(idx >> 7);
    if (n >= N) return;
    int j = (int)(idx & 127);
    float s = b1[j] + T[(size_t)ops[n] * EMB + j];
#pragma unroll
    for (int p = 0; p < PDIM; p++)
        s += T[(size_t)(1 + p) * 64 * EMB + (size_t)params[n * PDIM + p] * EMB + j];
    pre[idx] = s;
}

// ---------------------------------------------------------------------------
// WMMA matmul: OUT[16r x 128] = act( pre(X) @ W + b )
// PRE_MODE: 0 = X, 1 = lrelu(X), 2 = X + AGG/max(DEG,1)
// KT: number of 32-wide K chunks (4 -> K=128 from X1; 8 -> K=256 from X1|X2)
// NT_STORE: non-temporal output stores (write-once results)
template<int PRE_MODE, bool POST_LRELU, int KT, bool NT_STORE>
__global__ __launch_bounds__(256) void mm_wmma_kernel(
    const float* __restrict__ X1, const float* __restrict__ X2,
    const float* __restrict__ AGG, const float* __restrict__ DEG,
    const float* __restrict__ W, const float* __restrict__ Bb,
    float* __restrict__ OUT, int nRowTiles) {
    // Repack W (KT*32 x 128, row-major [k][n]) into B-operand layout:
    // sW[((kc*8+ct)*32 + lane)*16 + i] = W[kc*32 + (lane>>4)*16 + i][ct*16 + (lane&15)]
    __shared__ _Float16 sW[KT * 8 * 32 * 16];
    __shared__ float sB[EMB];
    for (int idx = threadIdx.x; idx < KT * 4096; idx += 256) {
        int i  = idx & 15;
        int ln = (idx >> 4) & 31;
        int ct = (idx >> 9) & 7;
        int kc = idx >> 12;
        int k = kc * 32 + (ln >> 4) * 16 + i;
        int n = ct * 16 + (ln & 15);
        sW[idx] = (_Float16)W[(size_t)k * EMB + n];
    }
    if (threadIdx.x < EMB) sB[threadIdx.x] = Bb[threadIdx.x];
    __syncthreads();

    int wave = threadIdx.x >> 5;
    int lane = threadIdx.x & 31;
    int tile = blockIdx.x * 8 + wave;
    if (tile >= nRowTiles) return;

    size_t rowBase = (size_t)tile * 16;
    int m  = lane & 15;
    int kh = lane >> 4;
    size_t xoff = (rowBase + m) * EMB;

    float inv = 1.0f;
    if (PRE_MODE == 2) inv = 1.0f / fmaxf(DEG[rowBase + m], 1.0f);

    v8f acc[8];
#pragma unroll
    for (int ct = 0; ct < 8; ct++)
#pragma unroll
        for (int r = 0; r < 8; r++) acc[ct][r] = 0.0f;

#pragma unroll
    for (int kc = 0; kc < KT; kc++) {
        const float* xp = ((KT == 8 && kc >= 4) ? X2 : X1) + xoff;
        int K0 = (kc & 3) * 32;
        // A operand (16x32 f16): lane<16: K = K0..K0+15; lane>=16: +16
        v16h a;
#pragma unroll
        for (int i = 0; i < 8; i++) {
            int k0 = K0 + kh * 16 + i;
            int k1 = K0 + kh * 16 + 8 + i;
            float v0 = xp[k0];
            float v1 = xp[k1];
            if (PRE_MODE == 1) { v0 = lrelu_f(v0); v1 = lrelu_f(v1); }
            if (PRE_MODE == 2) {
                const float* ap = AGG + xoff;
                v0 = __builtin_fmaf(ap[k0], inv, v0);
                v1 = __builtin_fmaf(ap[k1], inv, v1);
            }
            a[i]     = (_Float16)v0;
            a[8 + i] = (_Float16)v1;
        }
#pragma unroll
        for (int ct = 0; ct < 8; ct++) {
            v16h b = *(const v16h*)(sW + (((kc * 8 + ct) * 32 + lane) << 4));
            acc[ct] = __builtin_amdgcn_wmma_f32_16x16x32_f16(
                false, a, false, b, (short)0, acc[ct], false, false);
        }
    }

    // C/D layout: VGPR r -> M = r + (lane>=16 ? 8 : 0), N = lane&15
    int nlo = lane & 15;
    int mhi = (lane >> 4) * 8;
#pragma unroll
    for (int ct = 0; ct < 8; ct++) {
#pragma unroll
        for (int r = 0; r < 8; r++) {
            float v = acc[ct][r] + sB[ct * 16 + nlo];
            if (POST_LRELU) v = lrelu_f(v);
            float* dst = OUT + (rowBase + r + mhi) * EMB + ct * 16 + nlo;
            if (NT_STORE) __builtin_nontemporal_store(v, dst);
            else *dst = v;
        }
    }
}

// ---------------------------------------------------------------------------
// column sums (atomic accumulate into pre-zeroed out[128])
__global__ void colsum_kernel(const float* __restrict__ X, float* __restrict__ out, int n) {
    int col = threadIdx.x;  // 128 threads
    float s = 0.0f;
    for (int r = blockIdx.x; r < n; r += gridDim.x) s += X[(size_t)r * EMB + col];
    atomicAdd(out + col, s);
}

// ---------------------------------------------------------------------------
// graph-level MLPs: g_emb from conv_sums[0..2], g_inv from final_sum repeated 3x
__global__ void graph_mlp_kernel(const float* __restrict__ sums,  // 3*128, sums+256 = final
                                 const float* __restrict__ w1, const float* __restrict__ b1,
                                 const float* __restrict__ w2, const float* __restrict__ b2,
                                 const float* __restrict__ iw1, const float* __restrict__ ib1,
                                 const float* __restrict__ iw2, const float* __restrict__ ib2,
                                 float* __restrict__ gE, float* __restrict__ gI) {
    __shared__ float vin[384], vin2[384], t1[EMB], t2[EMB];
    int j = threadIdx.x;  // 128 threads
    for (int i = j; i < 384; i += EMB) {
        vin[i]  = sums[i];
        vin2[i] = sums[256 + (i & 127)];
    }
    __syncthreads();
    float s = b1[j], si = ib1[j];
    for (int k = 0; k < 384; k++) {
        s  += vin[k]  * w1[k * EMB + j];
        si += vin2[k] * iw1[k * EMB + j];
    }
    t1[j] = lrelu_f(s);
    t2[j] = lrelu_f(si);
    __syncthreads();
    float o = b2[j], oi = ib2[j];
    for (int k = 0; k < EMB; k++) {
        o  += t1[k] * w2[k * EMB + j];
        oi += t2[k] * iw2[k * EMB + j];
    }
    gE[j] = o;
    gI[j] = oi;
}

// out1[j] = comb_b[j] + [gE|gI](256) . comb_w[:,j]
__global__ void out1_kernel(const float* __restrict__ gE, const float* __restrict__ gI,
                            const float* __restrict__ cw, const float* __restrict__ cb,
                            float* __restrict__ out1) {
    int j = threadIdx.x;  // 128 threads
    float s = cb[j];
    for (int k = 0; k < EMB; k++) {
        s += gE[k] * cw[k * EMB + j];
        s += gI[k] * cw[(EMB + k) * EMB + j];
    }
    out1[j] = s;
}

// ---------------------------------------------------------------------------
extern "C" void kernel_launch(void* const* d_in, const int* in_sizes, int n_in,
                              void* d_out, int out_size, void* d_ws, size_t ws_size,
                              hipStream_t stream) {
    const int N = in_sizes[0];
    const int E = in_sizes[2];

    const int*   ops      = (const int*)d_in[0];
    const int*   params   = (const int*)d_in[1];
    const int*   esrc     = (const int*)d_in[2];
    const int*   edst     = (const int*)d_in[3];
    const float* op_tab   = (const float*)d_in[4];
    const float* par_tab  = (const float*)d_in[5];
    const float* om_w1    = (const float*)d_in[6];
    const float* om_b1    = (const float*)d_in[7];
    const float* om_w2    = (const float*)d_in[8];
    const float* om_b2    = (const float*)d_in[9];
    const float* conv_w   = (const float*)d_in[10];
    const float* conv_b   = (const float*)d_in[11];
    const float* mlp_w1   = (const float*)d_in[12];
    const float* mlp_b1   = (const float*)d_in[13];
    const float* mlp_w2   = (const float*)d_in[14];
    const float* mlp_b2   = (const float*)d_in[15];
    const float* imlp_w1  = (const float*)d_in[16];
    const float* imlp_b1  = (const float*)d_in[17];
    const float* imlp_w2  = (const float*)d_in[18];
    const float* imlp_b2  = (const float*)d_in[19];
    const float* comb_w   = (const float*)d_in[20];
    const float* comb_b   = (const float*)d_in[21];

    float* out1 = (float*)d_out;
    float* out2 = (float*)d_out + EMB;

    // workspace layout (floats)
    size_t NE = (size_t)N * EMB;
    float* ws   = (float*)d_ws;
    float* h    = ws;            // N*128
    float* sA   = h + NE;        // N*128
    float* sB   = sA + NE;       // N*128
    float* agg  = sB + NE;       // N*128 (also used as 'pre')
    float* degF = agg + NE;      // N
    float* degI = degF + N;      // N
    float* T    = degI + N;      // 5*64*128
    float* sums = T + 5 * 64 * EMB;  // 3*128
    float* gE   = sums + 3 * EMB;    // 128
    float* gI   = gE + EMB;          // 128

    const int nRowTiles = N / 16;
    const int mmBlocks  = (nRowTiles + 7) / 8;

    // degrees (once per direction) + conv-sum accumulator zero
    zero_kernel<<<(2 * N + 255) / 256, 256, 0, stream>>>(degF, 2L * N);
    zero_kernel<<<(3 * EMB + 255) / 256, 256, 0, stream>>>(sums, 3 * EMB);
    count_deg_kernel<<<(E + 255) / 256, 256, 0, stream>>>(edst, degF, E);
    count_deg_kernel<<<(E + 255) / 256, 256, 0, stream>>>(esrc, degI, E);

    // OpEmb: fold tables through opmlp_w1, gather, then h = lrelu(pre) @ W2 + b2
    precompute_T_kernel<<<(5 * 64 * EMB + 255) / 256, 256, 0, stream>>>(op_tab, par_tab, om_w1, T);
    gather_pre_kernel<<<(int)((NE + 255) / 256), 256, 0, stream>>>(ops, params, T, om_b1, agg, N);
    mm_wmma_kernel<1, false, 4, false><<<mmBlocks, 256, 0, stream>>>(
        agg, nullptr, nullptr, nullptr, om_w2, om_b2, h, nRowTiles);

    long eThreads = (long)E * 32;
    int  eBlocks  = (int)((eThreads + 255) / 256);

    // forward GIN layers: h -> sA -> sB -> sA
    const float* fIn[3]  = {h, sA, sB};
    float*       fOut[3] = {sA, sB, sA};
    for (int i = 0; i < 3; i++) {
        zero_kernel<<<(int)((NE + 255) / 256), 256, 0, stream>>>(agg, (long)NE);
        scatter_add_kernel<<<eBlocks, 256, 0, stream>>>(fIn[i], esrc, edst, agg, E);
        mm_wmma_kernel<2, true, 4, false><<<mmBlocks, 256, 0, stream>>>(
            fIn[i], nullptr, agg, degF, conv_w + (size_t)i * EMB * EMB, conv_b + i * EMB,
            fOut[i], nRowTiles);
        colsum_kernel<<<512, 128, 0, stream>>>(fOut[i], sums + i * EMB, N);
    }

    // inverse GIN layers (edges reversed): h -> sB -> h -> sB
    const float* iIn[3]  = {h, sB, h};
    float*       iOut[3] = {sB, h, sB};
    for (int i = 0; i < 3; i++) {
        zero_kernel<<<(int)((NE + 255) / 256), 256, 0, stream>>>(agg, (long)NE);
        scatter_add_kernel<<<eBlocks, 256, 0, stream>>>(iIn[i], edst, esrc, agg, E);
        mm_wmma_kernel<2, true, 4, false><<<mmBlocks, 256, 0, stream>>>(
            iIn[i], nullptr, agg, degI, conv_w + (size_t)i * EMB * EMB, conv_b + i * EMB,
            iOut[i], nRowTiles);
    }

    // graph-level outputs
    graph_mlp_kernel<<<1, 128, 0, stream>>>(sums, mlp_w1, mlp_b1, mlp_w2, mlp_b2,
                                            imlp_w1, imlp_b1, imlp_w2, imlp_b2, gE, gI);
    out1_kernel<<<1, 128, 0, stream>>>(gE, gI, comb_w, comb_b, out1);

    // out2 = [states_fwd | states_inv] (N x 256) @ comb_w + comb_b  (write-once -> NT stores)
    mm_wmma_kernel<0, false, 8, true><<<mmBlocks, 256, 0, stream>>>(
        sA, sB, nullptr, nullptr, comb_w, comb_b, out2, nRowTiles);
}
